// RelativePositionMultiHeadAttention_46720654246325
// MI455X (gfx1250) — matrix operand order repaired and checked by reading, not compiled, not run on hardware
//
#include <hip/hip_runtime.h>

// ---------------------------------------------------------------------------
// Types for CDNA5 WMMA (wave32): V_WMMA_F32_16X16X32_BF16
// ---------------------------------------------------------------------------
typedef __attribute__((ext_vector_type(16))) __bf16 bf16x16;
typedef __attribute__((ext_vector_type(8)))  float  f32x8;

union BFrag  { bf16x16 v; uint4 q[2]; };
union Pack16 { unsigned short s[16]; uint4 q[2]; };

__device__ __forceinline__ unsigned short f2bf(float f) {
  union { float f; unsigned int u; } c; c.f = f;
  unsigned int r = c.u + 0x7fffu + ((c.u >> 16) & 1u);   // round-to-nearest-even
  return (unsigned short)(r >> 16);
}

__device__ __forceinline__ f32x8 wmma_bf16(const BFrag& a, const BFrag& b, f32x8 c) {
  return __builtin_amdgcn_wmma_f32_16x16x32_bf16(false, a.v, false, b.v,
                                                 (short)0, c, false, false);
}

// Async memory->LDS copy (CDNA5): 16B per lane, tracked by ASYNCcnt.
// LDS dest address = low 32 bits of the generic pointer (wave-relative offset).
__device__ __forceinline__ void async_copy_b128(void* lds_dst, const void* gsrc) {
  unsigned int       laddr = (unsigned int)(unsigned long long)lds_dst;
  unsigned long long gaddr = (unsigned long long)gsrc;
  asm volatile("global_load_async_to_lds_b128 %0, %1, off"
               :
               : "v"(laddr), "v"(gaddr)
               : "memory");
}
__device__ __forceinline__ void async_wait0() {
  asm volatile("s_wait_asynccnt 0x0" ::: "memory");
}

__device__ __forceinline__ Pack16 cvt16(const float* __restrict__ src) {
  Pack16 p;
  float4 f0 = *(const float4*)(src + 0);
  float4 f1 = *(const float4*)(src + 4);
  float4 f2 = *(const float4*)(src + 8);
  float4 f3 = *(const float4*)(src + 12);
  p.s[0]=f2bf(f0.x);  p.s[1]=f2bf(f0.y);  p.s[2]=f2bf(f0.z);  p.s[3]=f2bf(f0.w);
  p.s[4]=f2bf(f1.x);  p.s[5]=f2bf(f1.y);  p.s[6]=f2bf(f1.z);  p.s[7]=f2bf(f1.w);
  p.s[8]=f2bf(f2.x);  p.s[9]=f2bf(f2.y);  p.s[10]=f2bf(f2.z); p.s[11]=f2bf(f2.w);
  p.s[12]=f2bf(f3.x); p.s[13]=f2bf(f3.y); p.s[14]=f2bf(f3.z); p.s[15]=f2bf(f3.w);
  return p;
}

// Problem constants
#define BSZ   4
#define LSEQ  1024
#define DMODEL 1024
#define NH    16
#define DKH   64

enum { MODE_Q = 0, MODE_K = 1, MODE_V = 2, MODE_OUT = 3 };

// ---------------------------------------------------------------------------
// C[M,N] = A[M,K] * W[N,K]^T + bias.  M=4096, N=K=1024.
// Block: 128x128 tile, 256 threads (8 waves), each wave 32x64 (2x4 WMMA tiles).
// MODE_Q/K/V: A is fp32 x, output scattered bf16 to [B][H][L][64] (K adds Er).
// MODE_OUT  : A is bf16 attention output (async-staged), output fp32 to d_out.
// ---------------------------------------------------------------------------
template<int MODE>
__global__ __launch_bounds__(256) void gemm_wT_kernel(
    const float* __restrict__ Af, const unsigned short* __restrict__ Ab,
    const float* __restrict__ W,  const float* __restrict__ bias,
    const float* __restrict__ Er,
    unsigned short* __restrict__ Cb, float* __restrict__ Cf)
{
  const int Kdim = 1024, Ndim = 1024;
  const int LDT = 48;                          // LDS row stride (elements)
  __shared__ unsigned short As[128 * 48];
  __shared__ unsigned short Ws[128 * 48];

  const int t     = threadIdx.x;
  const int lane  = t & 31;
  const int wave  = t >> 5;
  const int r15   = lane & 15;
  const int hb    = lane >> 4;
  const int mbase = (wave & 3) * 32;
  const int nbase = (wave >> 2) * 64;
  const int m0    = blockIdx.y * 128;
  const int n0    = blockIdx.x * 128;
  const int lr    = t >> 1;                    // 0..127 row in tile
  const int ls    = t & 1;                     // half of the 32-wide k slab

  const f32x8 zero8 = {0.f,0.f,0.f,0.f,0.f,0.f,0.f,0.f};
  f32x8 acc[2][4];
  for (int mi = 0; mi < 2; ++mi)
    for (int ni = 0; ni < 4; ++ni) acc[mi][ni] = zero8;

  for (int k0 = 0; k0 < Kdim; k0 += 32) {
    __syncthreads();
    // ---- stage A tile (convert fp32 -> bf16 if needed) ----
    if (MODE == MODE_OUT) {
      const unsigned short* src = Ab + (size_t)(m0 + lr) * Kdim + k0 + ls * 16;
      unsigned short* dst = &As[lr * LDT + ls * 16];
      async_copy_b128(dst,     src);          // ASYNCcnt-tracked, no VGPR data
      async_copy_b128(dst + 8, src + 8);
    } else {
      Pack16 p = cvt16(Af + (size_t)(m0 + lr) * Kdim + k0 + ls * 16);
      *(uint4*)&As[lr * LDT + ls * 16]     = p.q[0];
      *(uint4*)&As[lr * LDT + ls * 16 + 8] = p.q[1];
    }
    // ---- stage W tile (rows of W are columns of W^T: already [N][K]) ----
    {
      Pack16 p = cvt16(W + (size_t)(n0 + lr) * Kdim + k0 + ls * 16);
      *(uint4*)&Ws[lr * LDT + ls * 16]     = p.q[0];
      *(uint4*)&Ws[lr * LDT + ls * 16 + 8] = p.q[1];
    }
    if (MODE == MODE_OUT) async_wait0();
    __syncthreads();

    // ---- A fragments: lane r15 = row, chunks at k = hb*8 and 16+hb*8 ----
    BFrag af[2];
    #pragma unroll
    for (int mi = 0; mi < 2; ++mi) {
      const unsigned short* pA = &As[(mbase + mi * 16 + r15) * LDT + hb * 8];
      af[mi].q[0] = *(const uint4*)(pA);
      af[mi].q[1] = *(const uint4*)(pA + 16);
    }
    // ---- B fragments: lane r15 = N col, chunks at k = hb*8 and 16+hb*8 ----
    #pragma unroll
    for (int ni = 0; ni < 4; ++ni) {
      BFrag bfr;
      const unsigned short* pB = &Ws[(nbase + ni * 16 + r15) * LDT + hb * 8];
      bfr.q[0] = *(const uint4*)(pB);
      bfr.q[1] = *(const uint4*)(pB + 16);
      #pragma unroll
      for (int mi = 0; mi < 2; ++mi)
        acc[mi][ni] = wmma_bf16(af[mi], bfr, acc[mi][ni]);
    }
  }

  // ---- epilogue: C/D layout row = hb*8 + rr, col = lane&15 ----
  #pragma unroll
  for (int mi = 0; mi < 2; ++mi) {
    #pragma unroll
    for (int ni = 0; ni < 4; ++ni) {
      #pragma unroll
      for (int rr = 0; rr < 8; ++rr) {
        const int m = m0 + mbase + mi * 16 + hb * 8 + rr;
        const int n = n0 + nbase + ni * 16 + r15;
        float val = acc[mi][ni][rr] + bias[n];
        if (MODE == MODE_OUT) {
          Cf[(size_t)m * Ndim + n] = val;
        } else {
          const int b = m >> 10, l = m & 1023;
          const int h = n >> 6,  d = n & 63;
          if (MODE == MODE_K)                       // fold relative-pos term:
            val += Er[(4999 - l) * DKH + d];        // BD[i,j] = Q_i . Er[4999-j]
          Cb[((((size_t)b * NH + h) * LSEQ) + l) * DKH + d] = f2bf(val);
        }
      }
    }
  }
}

// ---------------------------------------------------------------------------
// Flash attention: O = softmax(Q K'^T / 8) V  per (b,h).
// Block = 256 threads (8 waves) handles 128 query rows; wave owns 16 rows.
// Loops over 16 key tiles of 64; online softmax; never materializes scores.
// Q and K' tiles staged with GLOBAL_LOAD_ASYNC_TO_LDS_B128 (ASYNCcnt path).
// ---------------------------------------------------------------------------
__global__ __launch_bounds__(256) void flash_attn_kernel(
    const unsigned short* __restrict__ Q,
    const unsigned short* __restrict__ Kp,
    const unsigned short* __restrict__ V,
    unsigned short* __restrict__ O)
{
  const int LDQ = 72;                         // padded LDS row stride
  __shared__ unsigned short Qs[128 * 72];     // [row][d]
  __shared__ unsigned short Ks[64 * 72];      // [key][d]   (= B layout for QK^T)
  __shared__ unsigned short Vs[64 * 72];      // [d][key]   (= B layout for PV)
  __shared__ unsigned short Ps[128 * 72];     // probs, bf16, per-wave slices

  const int t    = threadIdx.x;
  const int lane = t & 31;
  const int wave = t >> 5;
  const int r15  = lane & 15;
  const int hb   = lane >> 4;
  const int m0   = wave * 16;

  const int bh = blockIdx.y;                  // b*16 + h
  const int qt = blockIdx.x;                  // query tile (128 rows)
  const size_t base = (size_t)bh * LSEQ * DKH;

  // ---- stage Q tile once (async memory->LDS) ----
  {
    const int lr = t >> 1, ls = t & 1;
    const unsigned short* src = Q + base + (size_t)(qt * 128 + lr) * DKH + ls * 32;
    unsigned short* dst = &Qs[lr * LDQ + ls * 32];
    async_copy_b128(dst,      src);
    async_copy_b128(dst +  8, src +  8);
    async_copy_b128(dst + 16, src + 16);
    async_copy_b128(dst + 24, src + 24);
  }
  async_wait0();
  __syncthreads();

  // Q A-fragments live in registers for the whole kernel (dk=64 -> 2 k-chunks)
  BFrag qf[2];
  #pragma unroll
  for (int kc = 0; kc < 2; ++kc) {
    const unsigned short* p = &Qs[(m0 + r15) * LDQ + kc * 32 + hb * 8];
    qf[kc].q[0] = *(const uint4*)(p);
    qf[kc].q[1] = *(const uint4*)(p + 16);
  }

  const f32x8 zero8 = {0.f,0.f,0.f,0.f,0.f,0.f,0.f,0.f};
  f32x8 oacc[4];
  #pragma unroll
  for (int t4 = 0; t4 < 4; ++t4) oacc[t4] = zero8;
  float mrow[8], lrow[8];
  #pragma unroll
  for (int rr = 0; rr < 8; ++rr) { mrow[rr] = -3.0e38f; lrow[rr] = 0.f; }

  const int klr = t >> 2, kls = t & 3;

  for (int kt = 0; kt < LSEQ / 64; ++kt) {
    __syncthreads();
    // ---- stage K' tile [key][d] via async copy ----
    {
      const unsigned short* src = Kp + base + (size_t)(kt * 64 + klr) * DKH + kls * 16;
      unsigned short* dst = &Ks[klr * LDQ + kls * 16];
      async_copy_b128(dst,     src);
      async_copy_b128(dst + 8, src + 8);
    }
    // ---- stage V tile transposed -> [d][key] (manual: async can't transpose) ----
    {
      const unsigned short* src = V + base + (size_t)(kt * 64 + klr) * DKH + kls * 16;
      Pack16 p;
      p.q[0] = ((const uint4*)src)[0];
      p.q[1] = ((const uint4*)src)[1];
      #pragma unroll
      for (int e = 0; e < 16; ++e)
        Vs[(kls * 16 + e) * LDQ + klr] = p.s[e];
    }
    if (kt + 1 < LSEQ / 64) {   // global_prefetch_b8 of next tiles
      __builtin_prefetch(Kp + base + (size_t)((kt + 1) * 64 + klr) * DKH + kls * 16, 0, 1);
      __builtin_prefetch(V  + base + (size_t)((kt + 1) * 64 + klr) * DKH + kls * 16, 0, 1);
    }
    async_wait0();
    __syncthreads();

    // ---- S(16x64) = Q K'^T, scaled by 1/sqrt(64) ----
    f32x8 sacc[4];
    #pragma unroll
    for (int t4 = 0; t4 < 4; ++t4) {
      sacc[t4] = zero8;
      #pragma unroll
      for (int kc = 0; kc < 2; ++kc) {
        BFrag kb;
        const unsigned short* p = &Ks[(t4 * 16 + r15) * LDQ + kc * 32 + hb * 8];
        kb.q[0] = *(const uint4*)(p);
        kb.q[1] = *(const uint4*)(p + 16);
        sacc[t4] = wmma_bf16(qf[kc], kb, sacc[t4]);
      }
      sacc[t4] = sacc[t4] * 0.125f;
    }

    // ---- online softmax: row m = hb*8+rr spans the 16 lanes of one half ----
    float mnew[8], alpha[8], rs[8];
    #pragma unroll
    for (int rr = 0; rr < 8; ++rr) {
      float tm = fmaxf(fmaxf(sacc[0][rr], sacc[1][rr]),
                       fmaxf(sacc[2][rr], sacc[3][rr]));
      tm = fmaxf(tm, __shfl_xor(tm, 1));
      tm = fmaxf(tm, __shfl_xor(tm, 2));
      tm = fmaxf(tm, __shfl_xor(tm, 4));
      tm = fmaxf(tm, __shfl_xor(tm, 8));
      mnew[rr]  = fmaxf(mrow[rr], tm);
      alpha[rr] = __expf(mrow[rr] - mnew[rr]);
      mrow[rr]  = mnew[rr];
      rs[rr]    = 0.f;
    }
    #pragma unroll
    for (int t4 = 0; t4 < 4; ++t4)
      #pragma unroll
      for (int rr = 0; rr < 8; ++rr) {
        float pv = __expf(sacc[t4][rr] - mnew[rr]);
        sacc[t4][rr] = pv;
        rs[rr] += pv;
      }
    #pragma unroll
    for (int rr = 0; rr < 8; ++rr) {
      float s = rs[rr];
      s += __shfl_xor(s, 1);
      s += __shfl_xor(s, 2);
      s += __shfl_xor(s, 4);
      s += __shfl_xor(s, 8);
      lrow[rr] = lrow[rr] * alpha[rr] + s;
    }
    #pragma unroll
    for (int t4 = 0; t4 < 4; ++t4)
      #pragma unroll
      for (int rr = 0; rr < 8; ++rr)
        oacc[t4][rr] *= alpha[rr];

    // ---- re-layout P through LDS (C-layout regs -> A-operand layout) ----
    #pragma unroll
    for (int t4 = 0; t4 < 4; ++t4)
      #pragma unroll
      for (int rr = 0; rr < 8; ++rr)
        Ps[(m0 + hb * 8 + rr) * LDQ + t4 * 16 + r15] = f2bf(sacc[t4][rr]);

    BFrag pa[2];
    #pragma unroll
    for (int kc = 0; kc < 2; ++kc) {
      const unsigned short* p = &Ps[(m0 + r15) * LDQ + kc * 32 + hb * 8];
      pa[kc].q[0] = *(const uint4*)(p);
      pa[kc].q[1] = *(const uint4*)(p + 16);
    }
    // ---- O += P @ V ----
    #pragma unroll
    for (int t4 = 0; t4 < 4; ++t4) {
      #pragma unroll
      for (int kc = 0; kc < 2; ++kc) {
        BFrag vb;
        const unsigned short* p = &Vs[(t4 * 16 + r15) * LDQ + kc * 32 + hb * 8];
        vb.q[0] = *(const uint4*)(p);
        vb.q[1] = *(const uint4*)(p + 16);
        oacc[t4] = wmma_bf16(pa[kc], vb, oacc[t4]);
      }
    }
  }

  // ---- write O as bf16 in [B][L][H*64] layout (input of output projection) ----
  const int b = bh >> 4;
  const int h = bh & 15;
  #pragma unroll
  for (int t4 = 0; t4 < 4; ++t4)
    #pragma unroll
    for (int rr = 0; rr < 8; ++rr) {
      const int lg = qt * 128 + m0 + hb * 8 + rr;
      const float val = oacc[t4][rr] / lrow[rr];
      O[((size_t)b * LSEQ + lg) * DMODEL + h * DKH + t4 * 16 + r15] = f2bf(val);
    }
}

// ---------------------------------------------------------------------------
extern "C" void kernel_launch(void* const* d_in, const int* in_sizes, int n_in,
                              void* d_out, int out_size, void* d_ws, size_t ws_size,
                              hipStream_t stream) {
  const float* x   = (const float*)d_in[0];
  const float* W_q = (const float*)d_in[1];
  const float* b_q = (const float*)d_in[2];
  const float* W_k = (const float*)d_in[3];
  const float* b_k = (const float*)d_in[4];
  const float* W_v = (const float*)d_in[5];
  const float* b_v = (const float*)d_in[6];
  const float* W_o = (const float*)d_in[7];
  const float* b_o = (const float*)d_in[8];
  const float* Er  = (const float*)d_in[9];
  float* out = (float*)d_out;

  const size_t elems = (size_t)BSZ * NH * LSEQ * DKH;  // 4M bf16 elements each
  unsigned short* q_ws = (unsigned short*)d_ws;
  unsigned short* k_ws = q_ws + elems;
  unsigned short* v_ws = k_ws + elems;
  unsigned short* o_ws = v_ws + elems;

  dim3 gg(DMODEL / 128, (BSZ * LSEQ) / 128);   // (8, 32)
  gemm_wT_kernel<MODE_Q><<<gg, 256, 0, stream>>>(x, nullptr, W_q, b_q, nullptr, q_ws, nullptr);
  gemm_wT_kernel<MODE_K><<<gg, 256, 0, stream>>>(x, nullptr, W_k, b_k, Er,      k_ws, nullptr);
  gemm_wT_kernel<MODE_V><<<gg, 256, 0, stream>>>(x, nullptr, W_v, b_v, nullptr, v_ws, nullptr);

  dim3 ga(LSEQ / 128, BSZ * NH);               // (8, 64)
  flash_attn_kernel<<<ga, 256, 0, stream>>>(q_ws, k_ws, v_ws, o_ws);

  gemm_wT_kernel<MODE_OUT><<<gg, 256, 0, stream>>>(nullptr, o_ws, W_o, b_o, nullptr, nullptr, out);
}
